// GreaseArgPreTrainingHead_455266533930
// MI455X (gfx1250) — matrix operand order repaired and checked by reading, not compiled, not run on hardware
//
#include <hip/hip_runtime.h>
#include <hip/hip_bf16.h>

// ---------------- problem constants ----------------
#define Bc   2
#define Sc   512
#define Hc   768
#define Gc   256
#define Vc   32000
#define Nc   4096
#define Ec   4096
#define NGc  8
#define T1c  16
#define T2c  8
#define PERG 512

typedef __attribute__((ext_vector_type(16))) __bf16 v16bf;
typedef __attribute__((ext_vector_type(8)))  float  v8f;
typedef __attribute__((ext_vector_type(4)))  unsigned int u32x4;

// LLVM AMDGPU scheduling-group masks
#define SG_MFMA_WMMA 0x008
#define SG_VMEM_READ 0x020
#define SG_DS_READ   0x100
#define SG_DS_WRITE  0x200

// =====================================================================
// 1) index builder: top_mask==1 list (128), top_mask==2 list (64),
//    comment pair target/source rows (64 each)
// =====================================================================
__global__ void build_indices_kernel(const int* __restrict__ cpm,
                                     const int* __restrict__ topm,
                                     int* __restrict__ top1,   // [128]
                                     int* __restrict__ top2,   // [64]
                                     int* __restrict__ ptgt,   // [64]
                                     int* __restrict__ psrc)   // [64]
{
    if (threadIdx.x != 0 || blockIdx.x != 0) return;
    int c1 = 0, c2 = 0;
    for (int n = 0; n < Nc; ++n) {
        int m = topm[n];
        if (m == 1 && c1 < NGc * T1c) top1[c1++] = n;
        else if (m == 2 && c2 < NGc * T2c) top2[c2++] = n;
    }
    int cp = 0;
    for (int f = 0; f < Bc * Sc && cp < NGc * T2c; ++f) {
        int v = cpm[f];
        if (v >= 0) { ptgt[cp] = v + (f / Sc) * Sc; psrc[cp] = f; ++cp; }
    }
}

// =====================================================================
// 2) f32 -> bf16 convert (grid-stride)
// =====================================================================
__global__ void f32_to_bf16_kernel(const float* __restrict__ in,
                                   __hip_bfloat16* __restrict__ out, long n)
{
    long i = (long)blockIdx.x * blockDim.x + threadIdx.x;
    long stride = (long)gridDim.x * blockDim.x;
    for (; i < n; i += stride) out[i] = __float2bfloat16(in[i]);
}

// edge_feat[e, 0:256]   = good[src(e)], edge_feat[e, 256:512] = good[dst(e)]
__global__ void edge_feat_kernel(const float* __restrict__ good,
                                 const int* __restrict__ eidx,
                                 __hip_bfloat16* __restrict__ out)
{
    long i = (long)blockIdx.x * blockDim.x + threadIdx.x; // E*2G elems
    if (i >= (long)Ec * 2 * Gc) return;
    int e = (int)(i / (2 * Gc));
    int j = (int)(i % (2 * Gc));
    int node = (j < Gc) ? eidx[e] : eidx[Ec + e];
    int c    = (j < Gc) ? j : j - Gc;
    out[i] = __float2bfloat16(good[(long)node * Gc + c]);
}

// =====================================================================
// 3) generic NT bf16 WMMA GEMM:  C[M,Nn] = A[M,K] * Bm[Nn,K]^T + bias
//    block = 256 threads (8 waves); tile 128(M) x 64(N); K step 32.
//    B tile double-buffered in LDS; last K-step peeled; per-iteration
//    schedule pinned with sched_group_barrier:
//      VMEM_READ x3 -> DS_READ x4 -> WMMA -> DS_READ x2 -> WMMA
//      -> DS_READ x2 -> WMMA x2 -> DS_WRITE
//    requires M%128==0, Nn%64==0, K%32==0, K>=64 (true for all uses)
// =====================================================================
__global__ void gemm_nt_bf16_wmma_kernel(const __hip_bfloat16* __restrict__ A,
                                         const __hip_bfloat16* __restrict__ Bm,
                                         const float* __restrict__ bias,
                                         float* __restrict__ C,
                                         int M, int Nn, int K)
{
    __shared__ __align__(16) __hip_bfloat16 sB[2][64 * 32]; // 2 x 4KB, [col][k]

    const int tid   = threadIdx.x;
    const int wave  = tid >> 5;
    const int lane  = tid & 31;
    const int lrow  = lane & 15;   // row / col within 16
    const int lhalf = lane >> 4;   // 0|1

    const long nbase = (long)blockIdx.x * 64;
    const long mbase = (long)blockIdx.y * 128;
    const long mrow  = mbase + (long)wave * 16 + lrow;
    const __hip_bfloat16* Arow = A + mrow * (long)K;

    // staging coordinates: thread -> (col = tid/4, 8-half chunk = tid%4)
    const int scol = tid >> 2;
    const int sch  = tid & 3;
    const __hip_bfloat16* Bstage = Bm + (nbase + scol) * (long)K + sch * 8;
    __hip_bfloat16* sdst0 = &sB[0][scol * 32 + sch * 8];
    __hip_bfloat16* sdst1 = &sB[1][scol * 32 + sch * 8];

    v8f acc[4] = {};

    union frag { v16bf v; u32x4 u[2]; };

#define LOAD_BFRAG(dst, sBc, s)                                              \
    do {                                                                     \
        const __hip_bfloat16* bp_ = (sBc) + ((s) * 16 + lrow) * 32 + 8 * lhalf; \
        (dst).u[0] = *(const u32x4*)(bp_);                                   \
        (dst).u[1] = *(const u32x4*)(bp_ + 16);                              \
    } while (0)

#define WMMA_STEP(sBc)                                                       \
    do {                                                                     \
        frag f0, f1;                                                         \
        LOAD_BFRAG(f0, sBc, 0);                                              \
        LOAD_BFRAG(f1, sBc, 1);                                              \
        acc[0] = __builtin_amdgcn_wmma_f32_16x16x32_bf16(                    \
            false, af.v, false, f0.v, (short)0, acc[0], false, false);       \
        LOAD_BFRAG(f0, sBc, 2);                                              \
        acc[1] = __builtin_amdgcn_wmma_f32_16x16x32_bf16(                    \
            false, af.v, false, f1.v, (short)0, acc[1], false, false);       \
        LOAD_BFRAG(f1, sBc, 3);                                              \
        acc[2] = __builtin_amdgcn_wmma_f32_16x16x32_bf16(                    \
            false, af.v, false, f0.v, (short)0, acc[2], false, false);       \
        acc[3] = __builtin_amdgcn_wmma_f32_16x16x32_bf16(                    \
            false, af.v, false, f1.v, (short)0, acc[3], false, false);       \
    } while (0)

    // ---- prologue: stage K-step 0 into buffer 0 ----
    *(u32x4*)sdst0 = *(const u32x4*)(Bstage);
    __syncthreads();

    int buf = 0;
    int k0 = 0;
    // steady state: all but the last K step (staging unconditional)
    for (; k0 + 32 < K; k0 += 32) {
        frag af;
        af.u[0] = *(const u32x4*)(Arow + k0 + 8 * lhalf);
        af.u[1] = *(const u32x4*)(Arow + k0 + 16 + 8 * lhalf);
        // global prefetch of next B tile (consumed after the WMMA burst)
        u32x4 st = *(const u32x4*)(Bstage + (k0 + 32));

        const __hip_bfloat16* sBc = &sB[buf][0];
        WMMA_STEP(sBc);

        // stage next tile into the other buffer after this step's reads
        *(u32x4*)((buf == 0) ? sdst1 : sdst0) = st;

        // pin the pipeline shape for this iteration
        __builtin_amdgcn_sched_group_barrier(SG_VMEM_READ, 3, 0);
        __builtin_amdgcn_sched_group_barrier(SG_DS_READ,   4, 0);
        __builtin_amdgcn_sched_group_barrier(SG_MFMA_WMMA, 1, 0);
        __builtin_amdgcn_sched_group_barrier(SG_DS_READ,   2, 0);
        __builtin_amdgcn_sched_group_barrier(SG_MFMA_WMMA, 1, 0);
        __builtin_amdgcn_sched_group_barrier(SG_DS_READ,   2, 0);
        __builtin_amdgcn_sched_group_barrier(SG_MFMA_WMMA, 2, 0);
        __builtin_amdgcn_sched_group_barrier(SG_DS_WRITE,  1, 0);

        __syncthreads();
        buf ^= 1;
    }
    // peeled final K step (no staging)
    {
        frag af;
        af.u[0] = *(const u32x4*)(Arow + k0 + 8 * lhalf);
        af.u[1] = *(const u32x4*)(Arow + k0 + 16 + 8 * lhalf);
        const __hip_bfloat16* sBc = &sB[buf][0];
        WMMA_STEP(sBc);
    }
#undef WMMA_STEP
#undef LOAD_BFRAG

    // epilogue: VGPR v -> row v + 8*lhalf, col lrow (ISA f32 C/D layout)
#pragma unroll
    for (int s = 0; s < 4; ++s) {
#pragma unroll
        for (int v = 0; v < 8; ++v) {
            long m = mbase + (long)wave * 16 + v + 8 * lhalf;
            long n = nbase + s * 16 + lrow;
            C[m * (long)Nn + n] = acc[s][v] + (bias ? bias[n] : 0.0f);
        }
    }
}

// =====================================================================
// 4) exact GELU + LayerNorm(eps=1e-12); writes f32 and optional bf16
//    one block (256 thr) per row of 768
// =====================================================================
__global__ void gelu_ln_kernel(const float* __restrict__ pre,
                               const float* __restrict__ g,
                               const float* __restrict__ be,
                               float* __restrict__ xf,
                               __hip_bfloat16* __restrict__ xb)
{
    __shared__ float ssum[256], ssq[256];
    const long row = blockIdx.x;
    const int  t   = threadIdx.x;
    const float* p = pre + row * Hc;

    float y[3], s = 0.f, s2 = 0.f;
#pragma unroll
    for (int i = 0; i < 3; ++i) {
        float x = p[t + i * 256];
        float v = 0.5f * x * (1.0f + erff(x * 0.70710678118654752f));
        y[i] = v; s += v; s2 += v * v;
    }
    ssum[t] = s; ssq[t] = s2; __syncthreads();
    for (int off = 128; off > 0; off >>= 1) {
        if (t < off) { ssum[t] += ssum[t + off]; ssq[t] += ssq[t + off]; }
        __syncthreads();
    }
    const float mu  = ssum[0] * (1.0f / Hc);
    const float var = ssq[0] * (1.0f / Hc) - mu * mu;
    const float rs  = rsqrtf(var + 1e-12f);
#pragma unroll
    for (int i = 0; i < 3; ++i) {
        int h = t + i * 256;
        float o = (y[i] - mu) * rs * g[h] + be[h];
        if (xf) xf[row * Hc + h] = o;
        if (xb) xb[row * Hc + h] = __float2bfloat16(o);
    }
}

// =====================================================================
// 5) small heads
// =====================================================================
// top_x[r] = node_x[top1[r]] @ top_w^T + top_b ; one block per row
__global__ void top_x_kernel(const float* __restrict__ nodex,
                             const int* __restrict__ top1,
                             const float* __restrict__ topw,
                             const float* __restrict__ topb,
                             float* __restrict__ topx)
{
    __shared__ float sx[Hc];
    const int r = blockIdx.x, t = threadIdx.x;
    const int node = top1[r];
    for (int i = t; i < Hc; i += 256) sx[i] = nodex[(long)node * Hc + i];
    __syncthreads();
    for (int o = t; o < Hc; o += 256) {
        const float* w = topw + (long)o * Hc;
        float acc = topb[o];
        for (int k = 0; k < Hc; ++k) acc += sx[k] * w[k];
        topx[(long)r * Hc + o] = acc;
    }
}

// top_out[g,i,j] = dot(top_x[g,i], top_x[g,j]) ; block per graph, thread per (i,j)
__global__ void top_out_kernel(const float* __restrict__ topx, float* __restrict__ out)
{
    const int gph = blockIdx.x, t = threadIdx.x;
    const int i = t >> 4, j = t & 15;
    const float* a = topx + ((long)gph * T1c + i) * Hc;
    const float* b = topx + ((long)gph * T1c + j) * Hc;
    float acc = 0.f;
    for (int k = 0; k < Hc; ++k) acc += a[k] * b[k];
    out[gph * (T1c * T1c) + i * T1c + j] = acc;
}

// out[i] = dot(X[i], w) + b0 ; one thread per row
__global__ void rowdot_kernel(const float* __restrict__ X,
                              const float* __restrict__ w,
                              const float* __restrict__ b,
                              float* __restrict__ out, int n)
{
    int i = blockIdx.x * blockDim.x + threadIdx.x;
    if (i >= n) return;
    const float* x = X + (long)i * Hc;
    float acc = b[0];
    for (int k = 0; k < Hc; ++k) acc += x[k] * w[k];
    out[i] = acc;
}

// match_out[k] over concat(text_x[tgt], text_x[src], node_x[top2]) @ match_w^T + b
__global__ void match_kernel(const float* __restrict__ textx,
                             const float* __restrict__ nodex,
                             const int* __restrict__ ptgt,
                             const int* __restrict__ psrc,
                             const int* __restrict__ top2,
                             const float* __restrict__ mw,
                             const float* __restrict__ mb,
                             float* __restrict__ out)
{
    int k = threadIdx.x;
    if (k >= NGc * T2c) return;
    const float* t1 = textx + (long)ptgt[k] * Hc;
    const float* t2 = textx + (long)psrc[k] * Hc;
    const float* t3 = nodex + (long)top2[k] * Hc;
    float acc = mb[0];
    for (int h = 0; h < Hc; ++h)
        acc += t1[h] * mw[h] + t2[h] * mw[Hc + h] + t3[h] * mw[2 * Hc + h];
    out[k] = acc;
}

// =====================================================================
// launch
// =====================================================================
extern "C" void kernel_launch(void* const* d_in, const int* in_sizes, int n_in,
                              void* d_out, int out_size, void* d_ws, size_t ws_size,
                              hipStream_t stream)
{
    const float* lm_x    = (const float*)d_in[0];
    const float* goodx   = (const float*)d_in[1];
    const float* badx_in = (const float*)d_in[2];
    const int*   cpm     = (const int*)d_in[3];
    const int*   eidx    = (const int*)d_in[4];
    const int*   topm    = (const int*)d_in[5];
    /* d_in[6] = ptr, unused by reference */
    const float* text_w  = (const float*)d_in[7];
    const float* text_b  = (const float*)d_in[8];
    const float* text_g  = (const float*)d_in[9];
    const float* text_be = (const float*)d_in[10];
    const float* node_w  = (const float*)d_in[11];
    const float* node_b  = (const float*)d_in[12];
    const float* node_g  = (const float*)d_in[13];
    const float* node_be = (const float*)d_in[14];
    const float* edge_w  = (const float*)d_in[15];
    const float* edge_b  = (const float*)d_in[16];
    const float* edge_g  = (const float*)d_in[17];
    const float* edge_be = (const float*)d_in[18];
    const float* dec_w   = (const float*)d_in[19];
    const float* dec_b   = (const float*)d_in[20];
    const float* match_w = (const float*)d_in[21];
    const float* match_b = (const float*)d_in[22];
    const float* top_w   = (const float*)d_in[23];
    const float* top_b   = (const float*)d_in[24];
    const float* dir_w   = (const float*)d_in[25];
    const float* dir_b   = (const float*)d_in[26];
    const float* cont_w  = (const float*)d_in[27];
    const float* cont_b  = (const float*)d_in[28];

    float* out = (float*)d_out;
    const size_t OUT_TEXT = 0;
    const size_t OUT_NODE = OUT_TEXT + (size_t)Bc * Sc * Vc;        //  32,768,000
    const size_t OUT_TOP  = OUT_NODE + (size_t)Nc * Vc;             // 163,840,000
    const size_t OUT_CONT = OUT_TOP + (size_t)NGc * T1c * T1c;      // +2048
    const size_t OUT_EDGE = OUT_CONT + (size_t)2 * Nc;              // +8192
    const size_t OUT_DIR  = OUT_EDGE + (size_t)Ec * Vc;             // 294,922,240
    const size_t OUT_MATCH= OUT_DIR + (size_t)Ec;                   // 294,926,336

    // ---- workspace bump allocator ----
    char* wp = (char*)d_ws;
    auto alloc = [&](size_t bytes) -> void* {
        void* p = (void*)wp;
        wp += (bytes + 255) & ~(size_t)255;
        return p;
    };
    typedef __hip_bfloat16 bf16;
    bf16* decw_bf  = (bf16*)alloc((size_t)Vc * Hc * 2);
    bf16* textw_bf = (bf16*)alloc((size_t)Hc * Hc * 2);
    bf16* nodew_bf = (bf16*)alloc((size_t)Hc * Gc * 2);
    bf16* edgew_bf = (bf16*)alloc((size_t)Hc * 2 * Gc * 2);
    bf16* lm_bf    = (bf16*)alloc((size_t)Bc * Sc * Hc * 2);
    bf16* good_bf  = (bf16*)alloc((size_t)Nc * Gc * 2);
    bf16* bad_bf   = (bf16*)alloc((size_t)Nc * Gc * 2);
    bf16* efeat_bf = (bf16*)alloc((size_t)Ec * 2 * Gc * 2);
    bf16* textx_bf = (bf16*)alloc((size_t)Bc * Sc * Hc * 2);
    bf16* nodex_bf = (bf16*)alloc((size_t)Nc * Hc * 2);
    bf16* edgex_bf = (bf16*)alloc((size_t)Ec * Hc * 2);
    float* pre     = (float*)alloc((size_t)Nc * Hc * 4);     // reused per branch
    float* textx_f = (float*)alloc((size_t)Bc * Sc * Hc * 4);
    float* nodex_f = (float*)alloc((size_t)Nc * Hc * 4);
    float* badx_f  = (float*)alloc((size_t)Nc * Hc * 4);
    float* edgex_f = (float*)alloc((size_t)Ec * Hc * 4);
    float* topx_f  = (float*)alloc((size_t)NGc * T1c * Hc * 4);
    int*   top1    = (int*)alloc(128 * sizeof(int));
    int*   top2    = (int*)alloc(64 * sizeof(int));
    int*   ptgt    = (int*)alloc(64 * sizeof(int));
    int*   psrc    = (int*)alloc(64 * sizeof(int));

    // ---- 1) index lists ----
    build_indices_kernel<<<1, 1, 0, stream>>>(cpm, topm, top1, top2, ptgt, psrc);

    // ---- 2) bf16 conversions + edge-feature gather ----
    auto cvt = [&](const float* src, bf16* dst, long n) {
        int blocks = (int)((n + 255) / 256); if (blocks > 4096) blocks = 4096;
        f32_to_bf16_kernel<<<blocks, 256, 0, stream>>>(src, dst, n);
    };
    cvt(dec_w,   decw_bf,  (long)Vc * Hc);
    cvt(text_w,  textw_bf, (long)Hc * Hc);
    cvt(node_w,  nodew_bf, (long)Hc * Gc);
    cvt(edge_w,  edgew_bf, (long)Hc * 2 * Gc);
    cvt(lm_x,    lm_bf,    (long)Bc * Sc * Hc);
    cvt(goodx,   good_bf,  (long)Nc * Gc);
    cvt(badx_in, bad_bf,   (long)Nc * Gc);
    edge_feat_kernel<<<(Ec * 2 * Gc) / 256, 256, 0, stream>>>(goodx, eidx, efeat_bf);

    // ---- 3) transform pre-GEMMs (WMMA) + GELU/LN ----
    // text: [1024,768] = lm @ text_w^T
    gemm_nt_bf16_wmma_kernel<<<dim3(Hc / 64, (Bc * Sc) / 128), 256, 0, stream>>>(
        lm_bf, textw_bf, text_b, pre, Bc * Sc, Hc, Hc);
    gelu_ln_kernel<<<Bc * Sc, 256, 0, stream>>>(pre, text_g, text_be, textx_f, textx_bf);

    // good nodes: [4096,768] = good @ node_w^T (K=256)
    gemm_nt_bf16_wmma_kernel<<<dim3(Hc / 64, Nc / 128), 256, 0, stream>>>(
        good_bf, nodew_bf, node_b, pre, Nc, Hc, Gc);
    gelu_ln_kernel<<<Nc, 256, 0, stream>>>(pre, node_g, node_be, nodex_f, nodex_bf);

    // bad nodes (f32 only, feeds cont head)
    gemm_nt_bf16_wmma_kernel<<<dim3(Hc / 64, Nc / 128), 256, 0, stream>>>(
        bad_bf, nodew_bf, node_b, pre, Nc, Hc, Gc);
    gelu_ln_kernel<<<Nc, 256, 0, stream>>>(pre, node_g, node_be, badx_f, (bf16*)nullptr);

    // edges: [4096,768] = edge_feat @ edge_w^T (K=512)
    gemm_nt_bf16_wmma_kernel<<<dim3(Hc / 64, Ec / 128), 256, 0, stream>>>(
        efeat_bf, edgew_bf, edge_b, pre, Ec, Hc, 2 * Gc);
    gelu_ln_kernel<<<Ec, 256, 0, stream>>>(pre, edge_g, edge_be, edgex_f, edgex_bf);

    // ---- 4) decoder logits (the big WMMA GEMMs, write straight to d_out) ----
    gemm_nt_bf16_wmma_kernel<<<dim3(Vc / 64, (Bc * Sc) / 128), 256, 0, stream>>>(
        textx_bf, decw_bf, dec_b, out + OUT_TEXT, Bc * Sc, Vc, Hc);
    gemm_nt_bf16_wmma_kernel<<<dim3(Vc / 64, Nc / 128), 256, 0, stream>>>(
        nodex_bf, decw_bf, dec_b, out + OUT_NODE, Nc, Vc, Hc);
    gemm_nt_bf16_wmma_kernel<<<dim3(Vc / 64, Ec / 128), 256, 0, stream>>>(
        edgex_bf, decw_bf, dec_b, out + OUT_EDGE, Ec, Vc, Hc);

    // ---- 5) small heads ----
    top_x_kernel<<<NGc * T1c, 256, 0, stream>>>(nodex_f, top1, top_w, top_b, topx_f);
    top_out_kernel<<<NGc, T1c * T1c, 0, stream>>>(topx_f, out + OUT_TOP);

    rowdot_kernel<<<Nc / 256, 256, 0, stream>>>(nodex_f, cont_w, cont_b, out + OUT_CONT, Nc);
    rowdot_kernel<<<Nc / 256, 256, 0, stream>>>(badx_f,  cont_w, cont_b, out + OUT_CONT + Nc, Nc);
    rowdot_kernel<<<Ec / 256, 256, 0, stream>>>(edgex_f, dir_w,  dir_b,  out + OUT_DIR, Ec);

    match_kernel<<<1, 64, 0, stream>>>(textx_f, nodex_f, ptgt, psrc, top2,
                                       match_w, match_b, out + OUT_MATCH);
}
